// RepeatNet_48326972014792
// MI455X (gfx1250) — matrix operand order repaired and checked by reading, not compiled, not run on hardware
//
#include <hip/hip_runtime.h>
#include <hip/hip_bf16.h>
#include <cstddef>

#define B_  512
#define S_  128
#define V_  50000
#define E_  128
#define H_  512
#define HD_ 256
#define G3_ 768   // 3*HD

typedef __attribute__((ext_vector_type(16))) __bf16 v16bf;
typedef __attribute__((ext_vector_type(8)))  float  v8f;

__device__ __forceinline__ v8f wmma_bf16(v16bf a, v16bf b, v8f c) {
  // D = A(16x32 bf16) * B(32x16 bf16) + C(16x16 f32)
  return __builtin_amdgcn_wmma_f32_16x16x32_bf16(false, a, false, b, (short)0, c, false, false);
}

// ---- fragment loaders (ISA 7.12.2 layouts, wave32) --------------------------
// A 16x32 bf16: lane m=lane&15 holds row m; element e -> K = k0+e+(e>=8?8:0)+8g
__device__ __forceinline__ v16bf load_a_f32(const float* __restrict__ src, int lda,
                                            int m, int g, int k0) {
  v16bf a;
#pragma unroll
  for (int e = 0; e < 16; ++e) {
    int k = k0 + e + ((e >> 3) << 3) + 8 * g;
    a[e] = (__bf16)src[(size_t)m * lda + k];
  }
  return a;
}
__device__ __forceinline__ v16bf load_a_lds(const __bf16* src, int lda,
                                            int m, int g, int k0) {
  v16bf a;
#pragma unroll
  for (int e = 0; e < 16; ++e) {
    int k = k0 + e + ((e >> 3) << 3) + 8 * g;
    a[e] = src[m * lda + k];
  }
  return a;
}
// B fragment from repacked bf16 weights: block (kt,nt) holds 32 lanes x 16 halves
// contiguous; per-lane read is one aligned 32-byte vector.
__device__ __forceinline__ v16bf load_b_packed(const __bf16* __restrict__ Wp,
                                               int ntiles, int kt, int nt, int lane) {
  return *(const v16bf*)(Wp + (((size_t)kt * ntiles + nt) * 32 + lane) * 16);
}

__device__ __forceinline__ float sigmoidf_(float x) { return 1.0f / (1.0f + __expf(-x)); }

// ============================================================================
// Repack f32 row-major W[K,N] into fragment-linear bf16 blocks [kt][nt][lane][16].
// grid=(K/32, N/16), block=32. Element (lane,e): k=kt*32+e+(e>=8?8:0)+8*(lane>>4), n=nt*16+(lane&15).
// ============================================================================
__global__ __launch_bounds__(32) void repack_b_kernel(const float* __restrict__ W,
                                                      __bf16* __restrict__ out, int N) {
  const int kt = blockIdx.x, nt = blockIdx.y;
  const int lane = threadIdx.x, g = lane >> 4, n = lane & 15;
  const int ntiles = N / 16;
  __bf16* dst = out + (((size_t)kt * ntiles + nt) * 32 + lane) * 16;
  const float* src = W + (size_t)(kt * 32 + 8 * g) * N + nt * 16 + n;
#pragma unroll
  for (int e = 0; e < 16; ++e) dst[e] = (__bf16)src[(size_t)(e + ((e >> 3) << 3)) * N];
}

// ============================================================================
// Kernel 1: bidirectional GRU scan. grid=(B/16, 2 dirs), block=128 (4 waves).
// h kept in LDS (f32 + bf16). Each wave computes 4 gate triples per step from
// packed bf16 weights; z/cand staged through LDS to keep VGPR pressure low.
// ============================================================================
__global__ __launch_bounds__(128, 1) void gru_scan_kernel(
    const int* __restrict__ item_seq, const float* __restrict__ emb,
    const __bf16* __restrict__ Wx_f, const __bf16* __restrict__ Uh_f, const float* __restrict__ b_f,
    const __bf16* __restrict__ Wx_b, const __bf16* __restrict__ Uh_b, const float* __restrict__ b_b,
    float* __restrict__ out, float* __restrict__ state) {
  const int dir = blockIdx.y;
  const int b0  = blockIdx.x * 16;
  const __bf16* Wx  = dir ? Wx_b : Wx_f;
  const __bf16* Uh  = dir ? Uh_b : Uh_f;
  const float* bias = dir ? b_b  : b_f;

  __shared__ __bf16 x_bf[16][E_];   // 4 KB
  __shared__ __bf16 h_bf[16][HD_];  // 8 KB
  __shared__ float  h_f [16][HD_];  // 16 KB
  __shared__ float  z_l [16][HD_];  // 16 KB
  __shared__ float  c_l [16][HD_];  // 16 KB
  __shared__ int    mrow[16];

  const int tid  = threadIdx.x;
  const int wave = tid >> 5;
  const int lane = tid & 31;
  const int g    = lane >> 4;
  const int m    = lane & 15;

  for (int i = tid; i < 16 * HD_; i += 128) {
    (&h_f[0][0])[i]  = 0.0f;
    (&h_bf[0][0])[i] = (__bf16)0.0f;
  }
  __syncthreads();

  for (int t = 0; t < S_; ++t) {
    const int ts = dir ? (S_ - 1 - t) : t;
    if (tid < 16) mrow[tid] = item_seq[(b0 + tid) * S_ + ts];
    __syncthreads();
    for (int i = tid; i < 16 * E_; i += 128) {
      int r = i >> 7, c = i & (E_ - 1);
      x_bf[r][c] = (__bf16)emb[(size_t)mrow[r] * E_ + c];
    }
    __syncthreads();

#pragma unroll 1
    for (int tt = 0; tt < 4; ++tt) {
      const int ntz = wave * 4 + tt;   // z col tile (0..15); r at +16, h at +32
      v8f az = {}, ar = {}, ah = {};
#pragma unroll 1
      for (int kt = 0; kt < E_ / 32; ++kt) {  // x @ Wx (K=128)
        v16bf a = load_a_lds(&x_bf[0][0], E_, m, g, kt * 32);
        az = wmma_bf16(a, load_b_packed(Wx, 48, kt, ntz,      lane), az);
        ar = wmma_bf16(a, load_b_packed(Wx, 48, kt, ntz + 16, lane), ar);
        ah = wmma_bf16(a, load_b_packed(Wx, 48, kt, ntz + 32, lane), ah);
      }
      v8f azh = {}, arh = {}, ahh = {};
#pragma unroll 1
      for (int kt = 0; kt < HD_ / 32; ++kt) {  // h @ Uh (K=256)
        v16bf a = load_a_lds(&h_bf[0][0], HD_, m, g, kt * 32);
        azh = wmma_bf16(a, load_b_packed(Uh, 48, kt, ntz,      lane), azh);
        arh = wmma_bf16(a, load_b_packed(Uh, 48, kt, ntz + 16, lane), arh);
        ahh = wmma_bf16(a, load_b_packed(Uh, 48, kt, ntz + 32, lane), ahh);
      }
      const int c = ntz * 16 + m;
      const float b0z = bias[c],            b1z = bias[G3_ + c];
      const float b0r = bias[HD_ + c],      b1r = bias[G3_ + HD_ + c];
      const float b0h = bias[2 * HD_ + c],  b1h = bias[G3_ + 2 * HD_ + c];
#pragma unroll
      for (int j = 0; j < 8; ++j) {
        const int row = j + 8 * g;
        float z  = sigmoidf_(az[j] + b0z + azh[j] + b1z);
        float rr = sigmoidf_(ar[j] + b0r + arh[j] + b1r);
        z_l[row][c] = z;
        c_l[row][c] = tanhf(ah[j] + b0h + rr * (ahh[j] + b1h));
      }
    }
    __syncthreads();  // all h_bf reads + gate writes done
    for (int i = tid; i < 16 * HD_; i += 128) {
      const int r = i >> 8, c = i & (HD_ - 1);
      float hprev = h_f[r][c];
      float z = z_l[r][c];
      float hn = z * hprev + (1.0f - z) * c_l[r][c];
      bool  msk = (mrow[r] != 0);
      float hkeep = msk ? hn : hprev;
      h_f[r][c]  = hkeep;
      h_bf[r][c] = (__bf16)hkeep;
      out[((size_t)(b0 + r) * S_ + ts) * H_ + dir * HD_ + c] = msk ? hn : 0.0f;
    }
    __syncthreads();
  }
  for (int i = tid; i < 16 * HD_; i += 128) {
    int r = i >> 8, c = i & (HD_ - 1);
    state[(size_t)(b0 + r) * H_ + dir * HD_ + c] = h_f[r][c];
  }
}

// ============================================================================
// Kernel 2: qWq[h] = state @ Wq_h (packed). grid=(B/16,H/16,3), block=32.
// ============================================================================
__global__ __launch_bounds__(32, 1) void qproj_kernel(
    const float* __restrict__ state, const __bf16* __restrict__ Wqp,
    float* __restrict__ qWq) {
  const __bf16* Wq = Wqp + (size_t)blockIdx.z * H_ * H_;
  float* op = qWq + (size_t)blockIdx.z * B_ * H_;
  const int m0 = blockIdx.x * 16, nt = blockIdx.y;
  const int lane = threadIdx.x, g = lane >> 4, m = lane & 15;
  v8f acc = {};
#pragma unroll 1
  for (int kt = 0; kt < H_ / 32; ++kt) {
    v16bf a = load_a_f32(state + (size_t)m0 * H_, H_, m, g, kt * 32);
    acc = wmma_bf16(a, load_b_packed(Wq, H_ / 16, kt, nt, lane), acc);
  }
#pragma unroll
  for (int j = 0; j < 8; ++j)
    op[(size_t)(m0 + j + 8 * g) * H_ + nt * 16 + m] = acc[j];
}

// ============================================================================
// Kernel 3: fused e[b,s] = v . tanh(qWq[b] + out[b,s]@Wk), masked.
// grid=(B*S/16, 3), block=32. A fragments hoisted (reused by all 32 N tiles).
// ============================================================================
__global__ __launch_bounds__(32, 1) void attn_score_kernel(
    const float* __restrict__ out, const __bf16* __restrict__ Wkp,
    const float* __restrict__ v0, const float* __restrict__ v1, const float* __restrict__ v2,
    const float* __restrict__ qWq, const int* __restrict__ item_seq,
    float* __restrict__ esc) {
  const int head = blockIdx.y;
  const __bf16* Wk = Wkp + (size_t)head * H_ * H_;
  const float* vv = (head == 0) ? v0 : (head == 1) ? v1 : v2;
  const float* qW = qWq + (size_t)head * B_ * H_;
  float* ep = esc + (size_t)head * B_ * S_;

  const int rt = blockIdx.x;
  const int b  = (rt * 16) / S_;
  const int s0 = (rt * 16) % S_;
  const float* A = out + (size_t)rt * 16 * H_;
  const int lane = threadIdx.x, g = lane >> 4, m = lane & 15;

  v16bf af[H_ / 32];
#pragma unroll
  for (int kt = 0; kt < H_ / 32; ++kt) af[kt] = load_a_f32(A, H_, m, g, kt * 32);

  float esum[8];
#pragma unroll
  for (int j = 0; j < 8; ++j) esum[j] = 0.0f;

#pragma unroll 1
  for (int nt = 0; nt < H_ / 16; ++nt) {
    v8f acc = {};
#pragma unroll
    for (int kt = 0; kt < H_ / 32; ++kt)
      acc = wmma_bf16(af[kt], load_b_packed(Wk, H_ / 16, kt, nt, lane), acc);
    const float qv = qW[(size_t)b * H_ + nt * 16 + m];
    const float vn = vv[nt * 16 + m];
#pragma unroll
    for (int j = 0; j < 8; ++j) esum[j] += tanhf(acc[j] + qv) * vn;
  }
#pragma unroll
  for (int off = 1; off <= 8; off <<= 1)
#pragma unroll
    for (int j = 0; j < 8; ++j) esum[j] += __shfl_xor(esum[j], off);

  if (m == 0) {
#pragma unroll
    for (int j = 0; j < 8; ++j) {
      const int row = j + 8 * g;
      const int s = s0 + row;
      bool msk = (item_seq[b * S_ + s] != 0);
      ep[(size_t)b * S_ + s] = msk ? esum[j] : -1e9f;
    }
  }
}

// Kernel 4: masked softmax over S, in place. grid=(B,3), block=128.
__global__ __launch_bounds__(128) void softmax_kernel(float* __restrict__ esc) {
  float* row = esc + ((size_t)blockIdx.y * B_ + blockIdx.x) * S_;
  const int t = threadIdx.x;
  __shared__ float red[128];
  float x = row[t];
  red[t] = x; __syncthreads();
  for (int off = 64; off; off >>= 1) { if (t < off) red[t] = fmaxf(red[t], red[t + off]); __syncthreads(); }
  float mx = red[0]; __syncthreads();
  float e = __expf(x - mx);
  red[t] = e; __syncthreads();
  for (int off = 64; off; off >>= 1) { if (t < off) red[t] += red[t + off]; __syncthreads(); }
  row[t] = e / red[0];
}

// Kernel 5: feat[which] = sum_s a[b,s]*out[b,s,:]. grid=(B,2), block=128.
__global__ __launch_bounds__(128) void feat_kernel(
    const float* __restrict__ aprob, const float* __restrict__ out,
    float* __restrict__ feat) {
  const int which = blockIdx.y;          // 0 -> mode head (0), 1 -> explore head (2)
  const int head  = which ? 2 : 0;
  const int b = blockIdx.x;
  const float* a = aprob + ((size_t)head * B_ + b) * S_;
  float acc[4] = {0.f, 0.f, 0.f, 0.f};
  for (int s = 0; s < S_; ++s) {
    const float av = a[s];
    const float* o = out + ((size_t)b * S_ + s) * H_;
#pragma unroll
    for (int q = 0; q < 4; ++q) acc[q] += av * o[threadIdx.x + q * 128];
  }
#pragma unroll
  for (int q = 0; q < 4; ++q)
    feat[((size_t)which * B_ + b) * H_ + threadIdx.x + q * 128] = acc[q];
}

// Kernel 6: p_mode = softmax(feat_m @ Wm + bm). grid=(B), block=64.
__global__ __launch_bounds__(64) void mode_kernel(
    const float* __restrict__ feat, const float* __restrict__ Wm,
    const float* __restrict__ bm, float* __restrict__ p_mode) {
  const int b = blockIdx.x;
  const int col = threadIdx.x >> 5, lane = threadIdx.x & 31;
  const float* f = feat + (size_t)b * H_;
  float s = 0.f;
  for (int h = lane; h < H_; h += 32) s += f[h] * Wm[h * 2 + col];
#pragma unroll
  for (int off = 16; off; off >>= 1) s += __shfl_down(s, off);
  __shared__ float lg[2];
  if (lane == 0) lg[col] = s + bm[col];
  __syncthreads();
  if (threadIdx.x == 0) {
    float mx = fmaxf(lg[0], lg[1]);
    float e0 = __expf(lg[0] - mx), e1 = __expf(lg[1] - mx);
    float inv = 1.0f / (e0 + e1);
    p_mode[b * 2 + 0] = e0 * inv;
    p_mode[b * 2 + 1] = e1 * inv;
  }
}

__global__ void zero_kernel(float* __restrict__ p, size_t n) {
  size_t i = (size_t)blockIdx.x * blockDim.x + threadIdx.x;
  if (i < n) p[i] = 0.0f;
}

// Kernel 7: scatter p_mode1 * a_r into d_out. grid=(B), block=S.
__global__ __launch_bounds__(128) void scatter_kernel(
    const float* __restrict__ aprob, const int* __restrict__ item_seq,
    const float* __restrict__ p_mode, float* __restrict__ outp) {
  const int b = blockIdx.x, s = threadIdx.x;
  const int item = item_seq[b * S_ + s];
  const float w = p_mode[b * 2 + 1] * aprob[((size_t)1 * B_ + b) * S_ + s];
  atomicAdd(&outp[(size_t)b * V_ + item], w);
}

// Kernel 8: d_out += p_mode0 * (feat_e @ We + be). grid=(B/16, V/16), block=32.
__global__ __launch_bounds__(32, 1) void final_kernel(
    const float* __restrict__ featE, const __bf16* __restrict__ Wep,
    const float* __restrict__ be, const float* __restrict__ p_mode,
    float* __restrict__ outp) {
  const int m0 = blockIdx.x * 16, nt = blockIdx.y;
  const int lane = threadIdx.x, g = lane >> 4, m = lane & 15;
  v8f acc = {};
#pragma unroll 1
  for (int kt = 0; kt < H_ / 32; ++kt) {
    v16bf a = load_a_f32(featE + (size_t)m0 * H_, H_, m, g, kt * 32);
    acc = wmma_bf16(a, load_b_packed(Wep, V_ / 16, kt, nt, lane), acc);
  }
  const int n = nt * 16 + m;
  const float bev = be[n];
#pragma unroll
  for (int j = 0; j < 8; ++j) {
    const int row = m0 + j + 8 * g;
    const size_t idx = (size_t)row * V_ + n;
    outp[idx] = p_mode[row * 2] * (acc[j] + bev) + outp[idx];
  }
}

// ============================================================================
extern "C" void kernel_launch(void* const* d_in, const int* in_sizes, int n_in,
                              void* d_out, int out_size, void* d_ws, size_t ws_size,
                              hipStream_t stream) {
  (void)in_sizes; (void)n_in; (void)out_size; (void)ws_size;
  const int*   item_seq = (const int*)d_in[0];
  const float* emb    = (const float*)d_in[1];
  const float* gwf_Wx = (const float*)d_in[2];
  const float* gwf_Uh = (const float*)d_in[3];
  const float* gwf_b  = (const float*)d_in[4];
  const float* gwb_Wx = (const float*)d_in[5];
  const float* gwb_Uh = (const float*)d_in[6];
  const float* gwb_b  = (const float*)d_in[7];
  const float* Wq_m = (const float*)d_in[8],  *Wk_m = (const float*)d_in[9],  *v_m = (const float*)d_in[10];
  const float* Wq_r = (const float*)d_in[11], *Wk_r = (const float*)d_in[12], *v_r = (const float*)d_in[13];
  const float* Wq_e = (const float*)d_in[14], *Wk_e = (const float*)d_in[15], *v_e = (const float*)d_in[16];
  const float* We = (const float*)d_in[17], *be = (const float*)d_in[18];
  const float* Wm = (const float*)d_in[19], *bm = (const float*)d_in[20];
  float* outp = (float*)d_out;

  // f32 scratch (~141.3 MB), then bf16 repacked weights (~55.5 MB)
  float* ws    = (float*)d_ws;
  float* gru_o = ws;                                  // [B,S,H]
  float* state = gru_o + (size_t)B_ * S_ * H_;        // [B,H]
  float* qWq   = state + (size_t)B_ * H_;             // [3,B,H]
  float* esc   = qWq   + (size_t)3 * B_ * H_;         // [3,B,S] scores -> probs
  float* feat  = esc   + (size_t)3 * B_ * S_;         // [2,B,H] (0=mode,1=explore)
  float* pmode = feat  + (size_t)2 * B_ * H_;         // [B,2] (pad to 1024)
  __bf16* wsWxf = (__bf16*)(pmode + 1024);            // 128x768
  __bf16* wsUhf = wsWxf + (size_t)E_  * G3_;          // 256x768
  __bf16* wsWxb = wsUhf + (size_t)HD_ * G3_;
  __bf16* wsUhb = wsWxb + (size_t)E_  * G3_;
  __bf16* wsWq  = wsUhb + (size_t)HD_ * G3_;          // [3][512x512]
  __bf16* wsWk  = wsWq  + (size_t)3 * H_ * H_;        // [3][512x512]
  __bf16* wsWe  = wsWk  + (size_t)3 * H_ * H_;        // 512x50000

  // ---- repack all B-side weights into fragment-linear bf16 ----
  repack_b_kernel<<<dim3(E_ / 32,  G3_ / 16), 32, 0, stream>>>(gwf_Wx, wsWxf, G3_);
  repack_b_kernel<<<dim3(HD_ / 32, G3_ / 16), 32, 0, stream>>>(gwf_Uh, wsUhf, G3_);
  repack_b_kernel<<<dim3(E_ / 32,  G3_ / 16), 32, 0, stream>>>(gwb_Wx, wsWxb, G3_);
  repack_b_kernel<<<dim3(HD_ / 32, G3_ / 16), 32, 0, stream>>>(gwb_Uh, wsUhb, G3_);
  repack_b_kernel<<<dim3(H_ / 32, H_ / 16), 32, 0, stream>>>(Wq_m, wsWq,                      H_);
  repack_b_kernel<<<dim3(H_ / 32, H_ / 16), 32, 0, stream>>>(Wq_r, wsWq + (size_t)H_ * H_,    H_);
  repack_b_kernel<<<dim3(H_ / 32, H_ / 16), 32, 0, stream>>>(Wq_e, wsWq + (size_t)2 * H_ * H_, H_);
  repack_b_kernel<<<dim3(H_ / 32, H_ / 16), 32, 0, stream>>>(Wk_m, wsWk,                      H_);
  repack_b_kernel<<<dim3(H_ / 32, H_ / 16), 32, 0, stream>>>(Wk_r, wsWk + (size_t)H_ * H_,    H_);
  repack_b_kernel<<<dim3(H_ / 32, H_ / 16), 32, 0, stream>>>(Wk_e, wsWk + (size_t)2 * H_ * H_, H_);
  repack_b_kernel<<<dim3(H_ / 32, V_ / 16), 32, 0, stream>>>(We, wsWe, V_);

  // ---- network ----
  gru_scan_kernel<<<dim3(B_ / 16, 2), 128, 0, stream>>>(
      item_seq, emb, wsWxf, wsUhf, gwf_b, wsWxb, wsUhb, gwb_b, gru_o, state);

  qproj_kernel<<<dim3(B_ / 16, H_ / 16, 3), 32, 0, stream>>>(state, wsWq, qWq);

  attn_score_kernel<<<dim3(B_ * S_ / 16, 3), 32, 0, stream>>>(
      gru_o, wsWk, v_m, v_r, v_e, qWq, item_seq, esc);

  softmax_kernel<<<dim3(B_, 3), 128, 0, stream>>>(esc);

  feat_kernel<<<dim3(B_, 2), 128, 0, stream>>>(esc, gru_o, feat);

  mode_kernel<<<dim3(B_), 64, 0, stream>>>(feat, Wm, bm, pmode);

  const size_t nout = (size_t)B_ * V_;
  zero_kernel<<<dim3((unsigned)((nout + 255) / 256)), 256, 0, stream>>>(outp, nout);

  scatter_kernel<<<dim3(B_), S_, 0, stream>>>(esc, item_seq, pmode, outp);

  final_kernel<<<dim3(B_ / 16, V_ / 16), 32, 0, stream>>>(
      feat + (size_t)B_ * H_, wsWe, be, pmode, outp);
}
